// APPNPNet_41850161332535
// MI455X (gfx1250) — matrix operand order repaired and checked by reading, not compiled, not run on hardware
//
#include <hip/hip_runtime.h>

// APPNP on MI455X (gfx1250): WMMA f16 GEMMs + L2-resident scatter propagation.
#define NNODES 100000
#define NEDGES 3200000
#define IN_C   512
#define HID_C  256
#define OUT_C  64
#define KSTEPS 10
#define ALPHA  0.1f

typedef __attribute__((ext_vector_type(16))) _Float16 v16h;
typedef __attribute__((ext_vector_type(8)))  _Float16 v8h;
typedef __attribute__((ext_vector_type(2)))  _Float16 v2h;
typedef __attribute__((ext_vector_type(8)))  float    v8f;

// LDS row stride: 40 halfs = 80 B (16B-aligned for b128 loads; r*20 dwords mod 64
// distinct for r=0..15 -> conflict-free 16-row fragment reads).
#define LSTRIDE 40

// ---------------- WMMA fragment builders (CDNA5 ISA §7.12.2 layouts) --------
// A 16x32 f16: lane m=lane%16; lanes<16 hold K {0..7,16..23}, lanes>=16 hold
// K {8..15,24..31}. Both runs are 8 contiguous halfs -> two ds_load_b128.
__device__ __forceinline__ v16h frag_a_lds(const _Float16* As, int lane, int rowBase) {
  const int  m  = lane & 15;
  const bool hi = lane >= 16;
  const _Float16* rp = As + (rowBase + m) * LSTRIDE;
  const v8h lo8 = *(const v8h*)(rp + (hi ? 8 : 0));
  const v8h hi8 = *(const v8h*)(rp + (hi ? 24 : 16));
  return __builtin_shufflevector(lo8, hi8, 0,1,2,3,4,5,6,7,8,9,10,11,12,13,14,15);
}

// B 32x16 f16: col n=lane%16; lanes<16 hold K {0..7,16..23}, lanes>=16 hold K {8..15,24..31}.
// BsT is stored K-major ([col][k]) so each 8-half k run is contiguous.
__device__ __forceinline__ v16h frag_b_lds(const _Float16* BsT, int lane, int colBase) {
  const int n  = lane & 15;
  const int kb = (lane >= 16) ? 8 : 0;
  const _Float16* cp = BsT + (colBase + n) * LSTRIDE + kb;
  const v8h lo8 = *(const v8h*)(cp);
  const v8h hi8 = *(const v8h*)(cp + 16);
  return __builtin_shufflevector(lo8, hi8, 0,1,2,3,4,5,6,7,8,9,10,11,12,13,14,15);
}

// ---------------- GEMM1: h1 = relu(x @ W1 + b1), f32 in -> f16 out ----------
// Block 256 thr (8 waves); block tile 128(M) x 128(N); wave w owns rows 16w..16w+15
// and loops 8 column tiles; K staged 32 at a time.
__global__ __launch_bounds__(256) void gemm1_wmma(
    const float* __restrict__ x, const float* __restrict__ W1,
    const float* __restrict__ b1, _Float16* __restrict__ h1) {
  __shared__ __align__(16) _Float16 As [128 * LSTRIDE];   // 10240 B, [row][k]
  __shared__ __align__(16) _Float16 BsT[128 * LSTRIDE];   // 10240 B, [col][k]
  const int t = threadIdx.x, lane = t & 31, wave = t >> 5;
  const int rowBlock = blockIdx.x * 128;
  const int colBase  = blockIdx.y * 128;
  const bool full = (rowBlock + 128 <= NNODES);  // block-uniform fast path

  const v8f vzero = {};
  v8f acc[8];
#pragma unroll
  for (int ct = 0; ct < 8; ++ct) acc[ct] = vzero;

  for (int k0 = 0; k0 < IN_C; k0 += 32) {
    __syncthreads();
    // A: 128x32 f32 -> f16; float2 loads, packed b32 LDS stores (2048 pairs).
    if (full) {
#pragma unroll
      for (int it = 0; it < 8; ++it) {
        const int p  = t + it * 256;
        const int r  = p >> 4, kp = (p & 15) * 2;
        const float2 v = *(const float2*)(x + (size_t)(rowBlock + r) * IN_C + k0 + kp);
        v2h pk; pk[0] = (_Float16)v.x; pk[1] = (_Float16)v.y;
        *(v2h*)(As + r * LSTRIDE + kp) = pk;
      }
    } else {
#pragma unroll
      for (int it = 0; it < 8; ++it) {
        const int p  = t + it * 256;
        const int r  = p >> 4, kp = (p & 15) * 2;
        const int gr = rowBlock + r;
        float2 v = make_float2(0.f, 0.f);
        if (gr < NNODES) v = *(const float2*)(x + (size_t)gr * IN_C + k0 + kp);
        v2h pk; pk[0] = (_Float16)v.x; pk[1] = (_Float16)v.y;
        *(v2h*)(As + r * LSTRIDE + kp) = pk;
      }
    }
    // B: 32x128 f32 -> f16, transposed into [col][k] (4096 elems as float2).
#pragma unroll
    for (int it = 0; it < 8; ++it) {
      const int p  = t + it * 256;
      const int kr = p >> 6, c = (p & 63) * 2;
      const float2 v = *(const float2*)(W1 + (size_t)(k0 + kr) * HID_C + colBase + c);
      BsT[(c    ) * LSTRIDE + kr] = (_Float16)v.x;
      BsT[(c + 1) * LSTRIDE + kr] = (_Float16)v.y;
    }
    __syncthreads();
    const v16h af = frag_a_lds(As, lane, wave * 16);
#pragma unroll
    for (int ct = 0; ct < 8; ++ct) {
      const v16h bf = frag_b_lds(BsT, lane, ct * 16);
      acc[ct] = __builtin_amdgcn_wmma_f32_16x16x32_f16(
          false, af, false, bf, (short)0, acc[ct], false, false);
    }
  }
  // C/D layout: vgpr j holds row j + (lane>=16 ? 8 : 0), col lane%16.
  const int m0 = (lane >= 16) ? 8 : 0;
  const int n  = lane & 15;
#pragma unroll
  for (int ct = 0; ct < 8; ++ct) {
    const int   col  = colBase + ct * 16 + n;
    const float bias = b1[col];
#pragma unroll
    for (int j = 0; j < 8; ++j) {
      const int row = rowBlock + wave * 16 + j + m0;
      if (full || row < NNODES) {
        float v = acc[ct][j] + bias;
        v = v > 0.0f ? v : 0.0f;                 // ReLU
        h1[(size_t)row * HID_C + col] = (_Float16)v;
      }
    }
  }
}

// ---------------- GEMM2: h = h1 @ W2 + b2, f16 in -> f32 out ----------------
__global__ __launch_bounds__(256) void gemm2_wmma(
    const _Float16* __restrict__ h1, const float* __restrict__ W2,
    const float* __restrict__ b2, float* __restrict__ h) {
  __shared__ __align__(16) _Float16 As [128 * LSTRIDE];   // 10240 B
  __shared__ __align__(16) _Float16 BsT[ 64 * LSTRIDE];   //  5120 B
  const int t = threadIdx.x, lane = t & 31, wave = t >> 5;
  const int rowBlock = blockIdx.x * 128;
  const bool full = (rowBlock + 128 <= NNODES);  // block-uniform fast path

  const v8f vzero = {};
  v8f acc[4];
#pragma unroll
  for (int ct = 0; ct < 4; ++ct) acc[ct] = vzero;

  for (int k0 = 0; k0 < HID_C; k0 += 32) {
    __syncthreads();
    // A: 128x32 f16 copy, 8 halfs per chunk (b128 load + b128 store; 512 chunks).
    if (full) {
#pragma unroll
      for (int it = 0; it < 2; ++it) {
        const int c0 = t + it * 256;
        const int r  = c0 >> 2, koff = (c0 & 3) * 8;
        const v8h v  = *(const v8h*)(h1 + (size_t)(rowBlock + r) * HID_C + k0 + koff);
        *(v8h*)(As + r * LSTRIDE + koff) = v;
      }
    } else {
#pragma unroll
      for (int it = 0; it < 2; ++it) {
        const int c0 = t + it * 256;
        const int r  = c0 >> 2, koff = (c0 & 3) * 8;
        const int gr = rowBlock + r;
        v8h v = {};
        if (gr < NNODES) v = *(const v8h*)(h1 + (size_t)gr * HID_C + k0 + koff);
        *(v8h*)(As + r * LSTRIDE + koff) = v;
      }
    }
    // B: 32x64 f32 -> f16 transposed (1024 float2 pairs).
#pragma unroll
    for (int it = 0; it < 4; ++it) {
      const int p  = t + it * 256;
      const int kr = p >> 5, c = (p & 31) * 2;
      const float2 v = *(const float2*)(W2 + (size_t)(k0 + kr) * OUT_C + c);
      BsT[(c    ) * LSTRIDE + kr] = (_Float16)v.x;
      BsT[(c + 1) * LSTRIDE + kr] = (_Float16)v.y;
    }
    __syncthreads();
    const v16h af = frag_a_lds(As, lane, wave * 16);
#pragma unroll
    for (int ct = 0; ct < 4; ++ct) {
      const v16h bf = frag_b_lds(BsT, lane, ct * 16);
      acc[ct] = __builtin_amdgcn_wmma_f32_16x16x32_f16(
          false, af, false, bf, (short)0, acc[ct], false, false);
    }
  }
  const int m0 = (lane >= 16) ? 8 : 0;
  const int n  = lane & 15;
#pragma unroll
  for (int ct = 0; ct < 4; ++ct) {
    const int   col  = ct * 16 + n;
    const float bias = b2[col];
#pragma unroll
    for (int j = 0; j < 8; ++j) {
      const int row = rowBlock + wave * 16 + j + m0;
      if (full || row < NNODES)
        h[(size_t)row * OUT_C + col] = acc[ct][j] + bias;
    }
  }
}

// ---------------- degree / norm prep ---------------------------------------
__global__ void k_deg_init(float* deg) {
  const int i = blockIdx.x * blockDim.x + threadIdx.x;
  if (i < NNODES) deg[i] = 1.0f;                 // self-loop contribution
}
__global__ void k_deg(const int* __restrict__ ei, float* deg) {
  const int e = blockIdx.x * blockDim.x + threadIdx.x;
  if (e < NEDGES) atomicAdd(&deg[ei[NEDGES + e]], 1.0f);   // dst row
}
__global__ void k_dinv(float* deg) {
  const int i = blockIdx.x * blockDim.x + threadIdx.x;
  if (i < NNODES) deg[i] = rsqrtf(deg[i]);       // deg >= 1 always (self loop)
}
// Pre-fold (1-alpha)*dinv[s]*dinv[d] into one weight per edge; int32 indices.
__global__ void k_edgeprep(const int* __restrict__ ei, const float* __restrict__ dinv,
                           int* __restrict__ s32, int* __restrict__ d32,
                           float* __restrict__ w) {
  const int e = blockIdx.x * blockDim.x + threadIdx.x;
  if (e < NEDGES) {
    const int s = ei[e], d = ei[NEDGES + e];
    s32[e] = s; d32[e] = d;
    w[e] = (1.0f - ALPHA) * dinv[s] * dinv[d];
  }
}

// ---------------- APPNP propagation step -----------------------------------
// zn = alpha*h + (1-alpha)*dinv[i]^2*z  (self loop + teleport; writes every elem)
__global__ void k_self(const float* __restrict__ z, const float* __restrict__ h,
                       const float* __restrict__ dinv, float* __restrict__ zn) {
  const int idx = blockIdx.x * blockDim.x + threadIdx.x;
  if (idx < NNODES * OUT_C) {
    const float di = dinv[idx >> 6];
    zn[idx] = ALPHA * h[idx] + (1.0f - ALPHA) * di * di * z[idx];
  }
}
// One wave (32 lanes) per edge; each lane carries a float2 of the 64 channels.
__global__ void k_scatter(const float* __restrict__ z, const int* __restrict__ s32,
                          const int* __restrict__ d32, const float* __restrict__ w,
                          float* __restrict__ zn) {
  const int gid  = blockIdx.x * blockDim.x + threadIdx.x;
  const int e    = gid >> 5;
  const int lane = gid & 31;
  if (e >= NEDGES) return;
  const int   s  = s32[e], d = d32[e];
  const float wt = w[e];
  const float2 zv = *(const float2*)(z + (size_t)s * OUT_C + lane * 2);
  atomicAdd(zn + (size_t)d * OUT_C + lane * 2,     wt * zv.x);
  atomicAdd(zn + (size_t)d * OUT_C + lane * 2 + 1, wt * zv.y);
}

// ---------------- driver ----------------------------------------------------
extern "C" void kernel_launch(void* const* d_in, const int* in_sizes, int n_in,
                              void* d_out, int out_size, void* d_ws, size_t ws_size,
                              hipStream_t stream) {
  const float* x  = (const float*)d_in[0];
  const int*   ei = (const int*)d_in[1];   // edge_index [2,E] (int32 per harness)
  const float* W1 = (const float*)d_in[2];
  const float* b1 = (const float*)d_in[3];
  const float* W2 = (const float*)d_in[4];
  const float* b2 = (const float*)d_in[5];
  float* out = (float*)d_out;

  // Workspace carving (256B aligned). zA/zB overlap dead h1 buffer.
  char* ws = (char*)d_ws;
  size_t off = 0;
  auto carve = [&](size_t bytes) {
    void* p = ws + off;
    off += (bytes + 255) & ~(size_t)255;
    return p;
  };
  float*    zA  = (float*)carve((size_t)NNODES * OUT_C * 4);   // 25.6 MB
  float*    zB  = (float*)carve((size_t)NNODES * OUT_C * 4);   // 25.6 MB
  _Float16* h1  = (_Float16*)zA;   // 51.2 MB alias of zA+zB; dead before zA used
  float*    h   = (float*)carve((size_t)NNODES * OUT_C * 4);   // 25.6 MB
  float*    dinv= (float*)carve((size_t)NNODES * 4);           // deg -> dinv in place
  int*      s32 = (int*)carve((size_t)NEDGES * 4);
  int*      d32 = (int*)carve((size_t)NEDGES * 4);
  float*    wE  = (float*)carve((size_t)NEDGES * 4);

  const int TB = 256;
  // Norm prep
  k_deg_init<<<(NNODES + TB - 1) / TB, TB, 0, stream>>>(dinv);
  k_deg     <<<(NEDGES + TB - 1) / TB, TB, 0, stream>>>(ei, dinv);
  k_dinv    <<<(NNODES + TB - 1) / TB, TB, 0, stream>>>(dinv);
  k_edgeprep<<<(NEDGES + TB - 1) / TB, TB, 0, stream>>>(ei, dinv, s32, d32, wE);

  // MLP via WMMA
  const int MBLK = (NNODES + 127) / 128;         // 782
  gemm1_wmma<<<dim3(MBLK, HID_C / 128), TB, 0, stream>>>(x, W1, b1, h1);
  gemm2_wmma<<<dim3(MBLK), TB, 0, stream>>>(h1, W2, b2, h);

  // K-step propagation; z0 = h (no copy). Ping-pong zA/zB, last step -> d_out.
  const int selfBlocks    = (NNODES * OUT_C + TB - 1) / TB;
  const int scatterBlocks = (int)(((size_t)NEDGES * 32 + TB - 1) / TB);
  const float* zin = h;
  for (int k = 0; k < KSTEPS; ++k) {
    float* zout = (k == KSTEPS - 1) ? out : ((k & 1) ? zB : zA);
    k_self   <<<selfBlocks,    TB, 0, stream>>>(zin, h, dinv, zout);
    k_scatter<<<scatterBlocks, TB, 0, stream>>>(zin, s32, d32, wE, zout);
    zin = zout;
  }
}
